// ToySchizoBet_85177791414735
// MI455X (gfx1250) — matrix operand order repaired and checked by reading, not compiled, not run on hardware
//
#include <hip/hip_runtime.h>
#include <hip/hip_bf16.h>

typedef __attribute__((ext_vector_type(16))) _Float16 v16h;
typedef __attribute__((ext_vector_type(8)))  _Float16 v8h;
typedef __attribute__((ext_vector_type(8)))  float    v8f;

namespace {
constexpr int kB = 16, kT = 2048, kD = 128, kV = 42;
constexpr int kTileT = 16;
constexpr float kOmega = 0.5235987755982988f;           // 2*pi/12
constexpr float kPhase = 2048.0f * 0.5235987755982988f; // T * omega
// f16 weight scratch layout (element offsets)
constexpr int kW1T = 64 * 256;    // w1^T  [64][256]
constexpr int kWVT = 128 * 128;   // wv^T  [128][128]
constexpr int kMGT = 128 * 256;   // mg^T  [128][256]
constexpr int kHDT = 48 * 128;    // head^T padded [48][128]
}

// Build a 16-bit A fragment (16x32, M=lane&15) from a 128-half LDS row:
// j=0..15 -> K = 32kb + hi*16 + j
__device__ __forceinline__ v16h ldsA(const _Float16* rp, int hi) {
  union { v16h v; v8h h[2]; } u;
  u.h[0] = *reinterpret_cast<const v8h*>(rp + hi * 16);
  u.h[1] = *reinterpret_cast<const v8h*>(rp + hi * 16 + 8);
  return u.v;
}

__device__ __forceinline__ v8f wmma_f16(v16h a, v16h b, v8f c) {
  return __builtin_amdgcn_wmma_f32_16x16x32_f16(false, a, false, b, (short)0, c,
                                                false, false);
}

__device__ __forceinline__ float sigmoidf_(float x) {
  return 1.0f / (1.0f + __expf(-x));
}

// ---------------------------------------------------------------------------
// Prep: convert weights to f16, N-major (so B fragments are contiguous v16h).
// ---------------------------------------------------------------------------
__global__ void __launch_bounds__(256) prep_kernel(
    const float* __restrict__ w1, const float* __restrict__ wv,
    const float* __restrict__ mg, const float* __restrict__ hw,
    _Float16* __restrict__ ws) {
  const int idx = blockIdx.x * 256 + threadIdx.x;
  const int S1 = kW1T, S2 = S1 + kWVT, S3 = S2 + kMGT, S4 = S3 + kHDT;
  if (idx < S1) {                               // w1: [256][64] -> [64][256]
    int n = idx / 256, k = idx % 256;
    ws[idx] = (_Float16)w1[k * 64 + n];
  } else if (idx < S2) {                        // wv: [128][128] transposed
    int j = idx - S1; int n = j / 128, k = j % 128;
    ws[idx] = (_Float16)wv[k * 128 + n];
  } else if (idx < S3) {                        // mg: [256][128] -> [128][256]
    int j = idx - S2; int n = j / 256, k = j % 256;
    ws[idx] = (_Float16)mg[k * 128 + n];
  } else if (idx < S4) {                        // head: [128][42] -> [48][128] padded
    int j = idx - S3; int n = j / 128, k = j % 128;
    ws[idx] = (n < kV) ? (_Float16)hw[k * kV + n] : (_Float16)0.0f;
  }
}

// ---------------------------------------------------------------------------
// Fused model: one 16-token tile per workgroup, 8 wave32 per workgroup.
// ---------------------------------------------------------------------------
__global__ void __launch_bounds__(256) fused_kernel(
    const int* __restrict__ input_ids, const float* __restrict__ emb,
    const float* __restrict__ b1, const float* __restrict__ w2f,
    const float* __restrict__ b2, const float* __restrict__ wv_b,
    const float* __restrict__ mg_b, const float* __restrict__ ln_g,
    const float* __restrict__ ln_b, const float* __restrict__ head_b,
    const _Float16* __restrict__ ws, float* __restrict__ out) {
  __shared__ __align__(16) _Float16 x_lds[32][128];   // tokens t0-16 .. t0+15
  __shared__ float                  tau_lds[16][16];  // [token][window]
  __shared__ __align__(16) float    v_lds[32][128];   // V for all staged tokens
  __shared__ __align__(16) _Float16 msg_lds[16][128];
  __shared__ __align__(16) float    y_lds[16][128];
  __shared__ __align__(16) _Float16 yn_lds[16][128];

  const int tid   = threadIdx.x;
  const int wv_id = tid >> 5;
  const int lane  = tid & 31;
  const int ln16  = lane & 15;
  const int hi    = lane >> 4;

  const int batch = blockIdx.x / (kT / kTileT);
  const int t0    = (blockIdx.x % (kT / kTileT)) * kTileT;

  const _Float16* w1t = ws;
  const _Float16* wvt = ws + kW1T;
  const _Float16* mgt = ws + kW1T + kWVT;
  const _Float16* hdt = ws + kW1T + kWVT + kMGT;

  // ---- Step 0: gather 32 token embeddings -> f16 LDS -------------------
  {
    const int* ids = input_ids + batch * kT;
    for (int idx = tid; idx < 32 * 128; idx += 256) {
      const int row = idx >> 7, col = idx & 127;
      const int t = t0 - 16 + row;
      float v = 0.0f;
      if (t >= 0) v = emb[ids[t] * kD + col];
      x_lds[row][col] = (_Float16)v;
    }
  }
  __syncthreads();

  // ---- Step 1: gating MLP -> tau. Both of this wave's tokens advance
  //      through K together so each w1 B-fragment is loaded once and feeds
  //      two independent WMMA chains. --------------------------------------
  {
    const float b2v = b2[0];
    const int lt0 = wv_id * 2;
    const _Float16* xrow0 = &x_lds[16 + lt0][0];
    const _Float16* xrow1 = &x_lds[17 + lt0][0];
    const _Float16* nrow0 = &x_lds[15 + lt0 - ln16][0];  // A row m = ln16 = w
    const _Float16* nrow1 = &x_lds[16 + lt0 - ln16][0];
    v8f acc0[4], acc1[4];
    #pragma unroll
    for (int nt = 0; nt < 4; ++nt) {
      const float bv = b1[nt * 16 + ln16];
      #pragma unroll
      for (int r = 0; r < 8; ++r) { acc0[nt][r] = bv; acc1[nt][r] = bv; }
    }
    #pragma unroll
    for (int kb = 0; kb < 8; ++kb) {
      const _Float16* rp0 = (kb < 4) ? (xrow0 + 32 * kb) : (nrow0 + 32 * (kb - 4));
      const _Float16* rp1 = (kb < 4) ? (xrow1 + 32 * kb) : (nrow1 + 32 * (kb - 4));
      const v16h a0 = ldsA(rp0, hi);
      const v16h a1 = ldsA(rp1, hi);
      #pragma unroll
      for (int nt = 0; nt < 4; ++nt) {
        const v16h bf = *reinterpret_cast<const v16h*>(
            w1t + (nt * 16 + ln16) * 256 + kb * 32 + hi * 16);
        acc0[nt] = wmma_f16(a0, bf, acc0[nt]);
        acc1[nt] = wmma_f16(a1, bf, acc1[nt]);
      }
    }
    // SiLU then dot with w2, reduce across the 16 lanes of each half-wave
    auto tau_epilogue = [&](const v8f* acc, int lt) {
      float part[8];
      #pragma unroll
      for (int r = 0; r < 8; ++r) part[r] = 0.0f;
      #pragma unroll
      for (int nt = 0; nt < 4; ++nt) {
        const float w2v = w2f[nt * 16 + ln16];
        #pragma unroll
        for (int r = 0; r < 8; ++r) {
          const float h = acc[nt][r];
          part[r] += w2v * (h * sigmoidf_(h));
        }
      }
      #pragma unroll
      for (int m = 1; m < 16; m <<= 1) {
        #pragma unroll
        for (int r = 0; r < 8; ++r) part[r] += __shfl_xor(part[r], m, 32);
      }
      float tv = part[0];
      #pragma unroll
      for (int r = 1; r < 8; ++r) if (ln16 == r) tv = part[r];
      if (ln16 < 8) tau_lds[lt][hi * 8 + ln16] = sigmoidf_(tv + b2v);
    };
    tau_epilogue(acc0, lt0);
    tau_epilogue(acc1, lt0 + 1);
  }

  // ---- Step 2: V = X @ wv + b for all 32 staged rows -------------------
  #pragma unroll
  for (int rep = 0; rep < 2; ++rep) {
    const int tt = wv_id + rep * 8;     // 16 (mt,nt) tiles over 8 waves
    const int mt = tt >> 3, nt = tt & 7;
    const _Float16* arow = &x_lds[mt * 16 + ln16][0];
    v8f acc;
    const float bv = wv_b[nt * 16 + ln16];
    #pragma unroll
    for (int r = 0; r < 8; ++r) acc[r] = bv;
    #pragma unroll
    for (int kb = 0; kb < 4; ++kb) {
      const v16h a  = ldsA(arow + 32 * kb, hi);
      const v16h bf = *reinterpret_cast<const v16h*>(
          wvt + (nt * 16 + ln16) * 128 + kb * 32 + hi * 16);
      acc = wmma_f16(a, bf, acc);
    }
    #pragma unroll
    for (int r = 0; r < 8; ++r)
      v_lds[mt * 16 + hi * 8 + r][nt * 16 + ln16] = acc[r];
  }
  __syncthreads();

  // ---- Step 3: msg[t] = sum_w tau[t][w] * V[t-w-1] ---------------------
  {
    const int lt = tid >> 4;
    const int d0 = (tid & 15) * 8;
    float m[8];
    #pragma unroll
    for (int j = 0; j < 8; ++j) m[j] = 0.0f;
    #pragma unroll
    for (int w = 0; w < 16; ++w) {
      const float tv = tau_lds[lt][w];
      const float* vr = &v_lds[15 + lt - w][d0];
      #pragma unroll
      for (int j = 0; j < 8; ++j) m[j] += tv * vr[j];
    }
    #pragma unroll
    for (int j = 0; j < 8; ++j) msg_lds[lt][d0 + j] = (_Float16)m[j];
  }
  __syncthreads();

  // ---- Step 4: y = [x | msg] @ mg + b (one N-tile per wave) ------------
  {
    const int nt = wv_id;
    const _Float16* xrow = &x_lds[16 + ln16][0];  // A row m = ln16 = token
    const _Float16* mrow = &msg_lds[ln16][0];
    v8f acc;
    const float bv = mg_b[nt * 16 + ln16];
    #pragma unroll
    for (int r = 0; r < 8; ++r) acc[r] = bv;
    #pragma unroll
    for (int kb = 0; kb < 8; ++kb) {
      const _Float16* rp = (kb < 4) ? (xrow + 32 * kb) : (mrow + 32 * (kb - 4));
      const v16h a  = ldsA(rp, hi);
      const v16h bf = *reinterpret_cast<const v16h*>(
          mgt + (nt * 16 + ln16) * 256 + kb * 32 + hi * 16);
      acc = wmma_f16(a, bf, acc);
    }
    #pragma unroll
    for (int r = 0; r < 8; ++r)
      y_lds[hi * 8 + r][nt * 16 + ln16] = acc[r];
  }
  __syncthreads();

  // ---- Step 5: LayerNorm + periodic scale (2 rows per wave) ------------
  {
    const int row = wv_id * 2 + hi;
    const int d0  = ln16 * 8;
    float vals[8];
    float s = 0.0f, sq = 0.0f;
    #pragma unroll
    for (int j = 0; j < 8; ++j) {
      vals[j] = y_lds[row][d0 + j];
      s += vals[j];
      sq += vals[j] * vals[j];
    }
    #pragma unroll
    for (int m = 1; m < 16; m <<= 1) {
      s  += __shfl_xor(s, m, 32);
      sq += __shfl_xor(sq, m, 32);
    }
    const float mu  = s * (1.0f / 128.0f);
    const float var = sq * (1.0f / 128.0f) - mu * mu;
    const float rs  = rsqrtf(var + 1e-5f);
    const int   t   = t0 + row;
    const float sc  = 1.0f + 0.15f * __sinf((float)t * kOmega + kPhase);
    #pragma unroll
    for (int j = 0; j < 8; ++j) {
      const float g = ln_g[d0 + j], bb = ln_b[d0 + j];
      yn_lds[row][d0 + j] = (_Float16)(((vals[j] - mu) * rs * g + bb) * sc);
    }
  }
  __syncthreads();

  // ---- Step 6: logits = yn @ head + b (N padded 42->48, waves 0..2) ----
  if (wv_id < 3) {
    const int nt  = wv_id;
    const int col = nt * 16 + ln16;
    const _Float16* arow = &yn_lds[ln16][0];
    v8f acc;
    const float bv = (col < kV) ? head_b[col] : 0.0f;
    #pragma unroll
    for (int r = 0; r < 8; ++r) acc[r] = bv;
    #pragma unroll
    for (int kb = 0; kb < 4; ++kb) {
      const v16h a  = ldsA(arow + 32 * kb, hi);
      const v16h bf = *reinterpret_cast<const v16h*>(
          hdt + col * 128 + kb * 32 + hi * 16);
      acc = wmma_f16(a, bf, acc);
    }
    if (col < kV) {
      #pragma unroll
      for (int r = 0; r < 8; ++r) {
        const long t = (long)batch * kT + t0 + hi * 8 + r;
        out[t * kV + col] = acc[r];
      }
    }
  }
}

extern "C" void kernel_launch(void* const* d_in, const int* in_sizes, int n_in,
                              void* d_out, int out_size, void* d_ws, size_t ws_size,
                              hipStream_t stream) {
  const int*   ids  = (const int*)  d_in[0];
  const float* emb  = (const float*)d_in[1];
  const float* w1   = (const float*)d_in[2];
  const float* b1   = (const float*)d_in[3];
  const float* w2   = (const float*)d_in[4];
  const float* b2   = (const float*)d_in[5];
  const float* wv_w = (const float*)d_in[6];
  const float* wv_b = (const float*)d_in[7];
  const float* mg_w = (const float*)d_in[8];
  const float* mg_b = (const float*)d_in[9];
  const float* ln_g = (const float*)d_in[10];
  const float* ln_b = (const float*)d_in[11];
  const float* hw   = (const float*)d_in[12];
  const float* hb   = (const float*)d_in[13];
  _Float16* ws  = (_Float16*)d_ws;
  float*    out = (float*)d_out;

  const int prepN = kW1T + kWVT + kMGT + kHDT;
  prep_kernel<<<(prepN + 255) / 256, 256, 0, stream>>>(w1, wv_w, mg_w, hw, ws);
  fused_kernel<<<kB * (kT / kTileT), 256, 0, stream>>>(
      ids, emb, b1, w2, b2, wv_b, mg_b, ln_g, ln_b, hb, ws, out);
}
